// AFT_Full_47691316855062
// MI455X (gfx1250) — matrix operand order repaired and checked by reading, not compiled, not run on hardware
//
#include <hip/hip_runtime.h>
#include <hip/hip_bf16.h>

// ---------------------------------------------------------------------------
// AFT-Full for MI455X (gfx1250): bf16 WMMA + TDM + async global->LDS staging.
//   B=8, H=W=128, C=512, tokens M = B*H*W = 131072.
// ---------------------------------------------------------------------------

typedef __bf16 bf16;
typedef __attribute__((ext_vector_type(4)))  __bf16 v4bf;
typedef __attribute__((ext_vector_type(8)))  __bf16 v8bf;
typedef __attribute__((ext_vector_type(16))) __bf16 v16bf;
typedef __attribute__((ext_vector_type(8)))  float  v8f;
typedef __attribute__((ext_vector_type(4)))  unsigned int v4u;
typedef __attribute__((ext_vector_type(8)))  int v8i;
typedef __attribute__((ext_vector_type(4)))  int v4i;

#define C_      512
#define MTOT    131072          // 8*128*128 tokens
#define LDA_Q   520             // 512 + 8 pad (bf16): stride 260 dwords, 260%64=4
#define LDB_Q   72              // 64 + 8 pad : stride 36 dwords (matches TDM pad)
#define LDA_A   136             // 128 + 8 pad: stride 68 dwords

static __device__ __forceinline__ v8f vzero8() {
    v8f z = {0.f,0.f,0.f,0.f,0.f,0.f,0.f,0.f};
    return z;
}

static __device__ __forceinline__ v8f wmma_bf16(v16bf a, v16bf b, v8f c) {
    return __builtin_amdgcn_wmma_f32_16x16x32_bf16(
        false, a, false, b, (short)0, c, false, false);
}

// 32-bit LDS offset of a generic pointer that points into shared memory.
static __device__ __forceinline__ unsigned int lds_off_u32(const void* p) {
    return (unsigned int)(unsigned long long)
        (__attribute__((address_space(3))) const char*)p;
}

// Async global -> LDS copy of 16 bytes, tracked on ASYNCcnt.
static __device__ __forceinline__ void async_ld_b128(unsigned int lds_byte,
                                                     const void* gptr) {
    asm volatile("global_load_async_to_lds_b128 %0, %1, off"
                 :: "v"(lds_byte), "v"((unsigned long long)(size_t)gptr)
                 : "memory");
}

static __device__ __forceinline__ void wait_async0() {
    asm volatile("s_wait_asynccnt 0x0" ::: "memory");
}

// TDM: DMA a 128-row x 64-elem bf16 tile (row stride 512 elems in global)
// into LDS, hardware-padding +16B after every 128B row -> LDS stride LDB_Q.
// D# per CDNA5 ISA ch.8 (group0: count/lds/global/type, group1: dims+pad).
// This toolchain's builtin takes 6 args (g0..g3 + extended group + cpol).
static __device__ __forceinline__ void tdm_stage_B(const bf16* gtile,
                                                   unsigned int lds_byte) {
    unsigned long long ga = (unsigned long long)(size_t)gtile;
    v4u g0;
    g0[0] = 1u;                                   // count=1, user mode
    g0[1] = lds_byte;                             // lds_addr (bytes)
    g0[2] = (unsigned int)ga;                     // global_addr[31:0]
    g0[3] = (unsigned int)(ga >> 32) | (2u << 30);// global_addr[56:32] | type=2
    v8i g1;
    g1[0] = (int)((1u << 16)        // data_size = 2 bytes (bf16)
                | (1u << 20)        // pad_enable
                | (4u << 22)        // pad_interval: 32 dwords (=128B row)
                | (3u << 25));      // pad_amount: 4 dwords (=16B)
    g1[1] = (int)(512u << 16);      // tensor_dim0 = 512 (lo16 @ bits63:48)
    g1[2] = (int)(512u << 16);      // dim0 hi16=0 | tensor_dim1 = 512 lo16
    g1[3] = (int)(64u  << 16);      // dim1 hi16=0 | tile_dim0 = 64 elems
    g1[4] = 128;                    // tile_dim1 = 128 rows, tile_dim2 = 0
    g1[5] = 512;                    // tensor_dim0_stride[31:0] = 512 elems
    g1[6] = 0;                      // stride hi | dim1_stride lo
    g1[7] = 0;
    v4i g2 = {0,0,0,0};
    v4i g3 = {0,0,0,0};
    v8i g4 = {0,0,0,0,0,0,0,0};
    __builtin_amdgcn_tensor_load_to_lds(g0, g1, g2, g3, g4, 0);
}

// A fragment (16x32 bf16) per ISA layout: lanes 0-15 rows, K halves split on
// lane bit 4; two 16B ds reads.
static __device__ __forceinline__ v16bf frag_a(const bf16* t, int lda,
                                               int row0, int k0, int lane) {
    const bf16* p = t + (row0 + (lane & 15)) * lda + k0 + ((lane >> 4) << 3);
    union { v16bf v; v8bf h[2]; } u;
    u.h[0] = *(const v8bf*)(p);
    u.h[1] = *(const v8bf*)(p + 16);
    return u.v;
}

// B fragment (32x16 bf16) from N-major LDS tile: lane holds column n,
// K runs k0 + 8*(lane>>4) + [0,8) and k0 + 16 + 8*(lane>>4) + [0,8).
static __device__ __forceinline__ v16bf frag_b(const bf16* tT, int ldb,
                                               int n0, int k0, int lane) {
    const bf16* p = tT + (n0 + (lane & 15)) * ldb + k0 + ((lane >> 4) << 3);
    union { v16bf v; v8bf h[2]; } u;
    u.h[0] = *(const v8bf*)(p);
    u.h[1] = *(const v8bf*)(p + 16);
    return u.v;
}

// Full-K (512) GEMM pass: A tile resident in LDS; W^T chunks double-buffered
// through LDS by the Tensor Data Mover (wave 0 issues, all gate on TENSORcnt).
static __device__ __forceinline__ void gemm_pass512(
        const bf16* At, bf16* Bt0, bf16* Bt1, const bf16* Wt, int N0,
        int lane, int wave, int wm, int wn, v8f acc[4][2]) {
    const bf16* Wtile = Wt + (((size_t)N0) << 9);
    if (wave == 0)
        tdm_stage_B(Wtile, lds_off_u32(Bt0));
    for (int kc = 0; kc < 512; kc += 64) {
        bf16* cur = ((kc >> 6) & 1) ? Bt1 : Bt0;
        bf16* nxt = ((kc >> 6) & 1) ? Bt0 : Bt1;
        wait_async0();                          // A-tile stage (1st iter only)
        __builtin_amdgcn_s_wait_tensorcnt(0);   // cur chunk DMA complete
        __syncthreads();
        if (wave == 0 && kc + 64 < 512)
            tdm_stage_B(Wtile + kc + 64, lds_off_u32(nxt));  // overlap w/ math
#pragma unroll
        for (int kk = 0; kk < 64; kk += 32) {
            v16bf b0 = frag_b(cur, LDB_Q, wn,      kk, lane);
            v16bf b1 = frag_b(cur, LDB_Q, wn + 16, kk, lane);
#pragma unroll
            for (int tm = 0; tm < 4; ++tm) {
                v16bf a = frag_a(At, LDA_Q, wm + 16 * tm, kc + kk, lane);
                acc[tm][0] = wmma_bf16(a, b0, acc[tm][0]);
                acc[tm][1] = wmma_bf16(a, b1, acc[tm][1]);
            }
        }
    }
}

// ---------------------------------------------------------------------------
// Kernel 1: x f32 -> bf16
// ---------------------------------------------------------------------------
__global__ __launch_bounds__(256) void cvt_x_kernel(const float* __restrict__ x,
                                                    bf16* __restrict__ xb, int n4) {
    int stride = gridDim.x * blockDim.x;
    for (int i = blockIdx.x * blockDim.x + threadIdx.x; i < n4; i += stride) {
        float4 f = ((const float4*)x)[i];
        v4bf o;
        o.x = (bf16)f.x; o.y = (bf16)f.y; o.z = (bf16)f.z; o.w = (bf16)f.w;
        ((v4bf*)xb)[i] = o;
    }
}

// ---------------------------------------------------------------------------
// Kernel 2: weights f32 -> bf16 transposed (WT[n*512+k] = W[k*512+n]),
//           order q,k,v,o; plus ew = exp(w[:128,:128]) -> bf16.
// ---------------------------------------------------------------------------
__global__ __launch_bounds__(256) void cvt_w_kernel(
        const float* __restrict__ Wq, const float* __restrict__ Wk,
        const float* __restrict__ Wv, const float* __restrict__ Wo,
        const float* __restrict__ wtab, bf16* __restrict__ WT,
        bf16* __restrict__ ewb) {
    const int total = 4 * 262144 + 16384;
    int stride = gridDim.x * blockDim.x;
    for (int i = blockIdx.x * blockDim.x + threadIdx.x; i < total; i += stride) {
        if (i < 1048576) {
            int sel = i >> 18;
            int rem = i & 262143;
            int k = rem >> 9, n = rem & 511;
            const float* src = (sel == 0) ? Wq : (sel == 1) ? Wk
                             : (sel == 2) ? Wv : Wo;
            WT[sel * 262144 + n * 512 + k] = (bf16)src[k * 512 + n];
        } else {
            int j = i - 1048576;
            int r = j >> 7, c = j & 127;
            ewb[j] = (bf16)__expf(wtab[r * 1024 + c]);   // w row stride MAX_LEN
        }
    }
}

// ---------------------------------------------------------------------------
// Kernel 3: fused QKV. 128x128 tile per block; A (x) tile async-staged once,
// resident across passes K -> V -> Q; exp(k) kept in registers for the V pass.
// ---------------------------------------------------------------------------
__global__ __launch_bounds__(256) void qkv_kernel(
        const bf16* __restrict__ xb,
        const bf16* __restrict__ WkT, const bf16* __restrict__ WvT,
        const bf16* __restrict__ WqT,
        const float* __restrict__ bk, const float* __restrict__ bv,
        const float* __restrict__ bq,
        bf16* __restrict__ ekb, bf16* __restrict__ ekvb,
        bf16* __restrict__ sqb) {
    extern __shared__ char smem[];
    bf16* At  = (bf16*)smem;               // 128 x LDA_Q
    bf16* Bt0 = At + 128 * LDA_Q;          // 128 x LDB_Q, double buffered
    bf16* Bt1 = Bt0 + 128 * LDB_Q;

    const int tid = threadIdx.x, lane = tid & 31, wave = tid >> 5;
    const int wm = (wave >> 2) * 64, wn = (wave & 3) * 32;
    const int N0 = blockIdx.x * 128;
    const size_t M0 = (size_t)blockIdx.y * 128;

    // async-stage A: 128 rows x 512 K bf16 (64 x 16B per row)
    for (int i = tid; i < 8192; i += 256) {
        int r = i >> 6, c = i & 63;
        async_ld_b128(lds_off_u32(At + r * LDA_Q + c * 8),
                      xb + ((M0 + r) << 9) + c * 8);
    }

    const int mo = (lane >> 4) << 3, no = lane & 15;
    v8f acc[4][2];
    float ekr[4][2][8];

    // ---- pass 1: K ----
#pragma unroll
    for (int tm = 0; tm < 4; ++tm) { acc[tm][0] = vzero8(); acc[tm][1] = vzero8(); }
    gemm_pass512(At, Bt0, Bt1, WkT, N0, lane, wave, wm, wn, acc);
#pragma unroll
    for (int tm = 0; tm < 4; ++tm)
#pragma unroll
        for (int tn = 0; tn < 2; ++tn) {
            const size_t col = (size_t)(N0 + wn + 16 * tn + no);
            const float bias = bk[col];
#pragma unroll
            for (int e = 0; e < 8; ++e) {
                size_t row = M0 + wm + 16 * tm + mo + e;
                float ek = __expf(acc[tm][tn][e] + bias);
                ekr[tm][tn][e] = ek;
                ekb[(row << 9) + col] = (bf16)ek;
            }
        }

    // ---- pass 2: V  (emit ek*v) ----
#pragma unroll
    for (int tm = 0; tm < 4; ++tm) { acc[tm][0] = vzero8(); acc[tm][1] = vzero8(); }
    gemm_pass512(At, Bt0, Bt1, WvT, N0, lane, wave, wm, wn, acc);
#pragma unroll
    for (int tm = 0; tm < 4; ++tm)
#pragma unroll
        for (int tn = 0; tn < 2; ++tn) {
            const size_t col = (size_t)(N0 + wn + 16 * tn + no);
            const float bias = bv[col];
#pragma unroll
            for (int e = 0; e < 8; ++e) {
                size_t row = M0 + wm + 16 * tm + mo + e;
                ekvb[(row << 9) + col] =
                    (bf16)(ekr[tm][tn][e] * (acc[tm][tn][e] + bias));
            }
        }

    // ---- pass 3: Q  (emit sigmoid(q)) ----
#pragma unroll
    for (int tm = 0; tm < 4; ++tm) { acc[tm][0] = vzero8(); acc[tm][1] = vzero8(); }
    gemm_pass512(At, Bt0, Bt1, WqT, N0, lane, wave, wm, wn, acc);
#pragma unroll
    for (int tm = 0; tm < 4; ++tm)
#pragma unroll
        for (int tn = 0; tn < 2; ++tn) {
            const size_t col = (size_t)(N0 + wn + 16 * tn + no);
            const float bias = bq[col];
#pragma unroll
            for (int e = 0; e < 8; ++e) {
                size_t row = M0 + wm + 16 * tm + mo + e;
                float s = 1.0f / (1.0f + __expf(-(acc[tm][tn][e] + bias)));
                sqb[(row << 9) + col] = (bf16)s;
            }
        }
}

// ---------------------------------------------------------------------------
// Kernel 4: per (b,h) batch:  num = ew @ ekv,  den = ew @ ek,
//           y = sigmoid(q) * num / den.  Grid: (C/128, 1024 batches).
// ---------------------------------------------------------------------------
__global__ __launch_bounds__(256) void aft_kernel(
        const bf16* __restrict__ ewb,
        const bf16* __restrict__ ekb, const bf16* __restrict__ ekvb,
        const bf16* __restrict__ sqb, bf16* __restrict__ yb) {
    extern __shared__ char smem[];
    bf16* Ew   = (bf16*)smem;                 // 128 x LDA_A (A operand)
    bf16* EkT  = Ew  + 128 * LDA_A;           // 128(N) x LDA_A(K), transposed
    bf16* EkvT = EkT + 128 * LDA_A;

    const int tid = threadIdx.x, lane = tid & 31, wave = tid >> 5;
    const int wm = (wave >> 2) * 64, wn = (wave & 3) * 32;
    const int N0 = blockIdx.x * 128;
    const size_t bh = blockIdx.y;

    const bf16* ekB  = ekb  + (bh << 16);     // 128*512 elems per batch
    const bf16* ekvB = ekvb + (bh << 16);

    // async-stage ew (row-major, contiguous 16B chunks)
    for (int i = tid; i < 2048; i += 256) {
        int r = i >> 4, c = i & 15;
        async_ld_b128(lds_off_u32(Ew + r * LDA_A + c * 8),
                      ewb + r * 128 + c * 8);
    }
    // stage ek / ekv transposed: LDS[n][k] = g[k][N0+n] (2B gather/scatter)
    for (int i = tid; i < 16384; i += 256) {
        int wr = i >> 7, c = i & 127;
        size_t g = (((size_t)wr) << 9) + N0 + c;
        EkT [c * LDA_A + wr] = ekB[g];
        EkvT[c * LDA_A + wr] = ekvB[g];
    }
    wait_async0();
    __syncthreads();

    v8f accn[4][2], accd[4][2];
#pragma unroll
    for (int tm = 0; tm < 4; ++tm) {
        accn[tm][0] = vzero8(); accn[tm][1] = vzero8();
        accd[tm][0] = vzero8(); accd[tm][1] = vzero8();
    }
#pragma unroll
    for (int kk = 0; kk < 128; kk += 32) {
        v16bf bn0 = frag_b(EkvT, LDA_A, wn,      kk, lane);
        v16bf bn1 = frag_b(EkvT, LDA_A, wn + 16, kk, lane);
        v16bf bd0 = frag_b(EkT,  LDA_A, wn,      kk, lane);
        v16bf bd1 = frag_b(EkT,  LDA_A, wn + 16, kk, lane);
#pragma unroll
        for (int tm = 0; tm < 4; ++tm) {
            v16bf a = frag_a(Ew, LDA_A, wm + 16 * tm, kk, lane);
            accn[tm][0] = wmma_bf16(a, bn0, accn[tm][0]);
            accn[tm][1] = wmma_bf16(a, bn1, accn[tm][1]);
            accd[tm][0] = wmma_bf16(a, bd0, accd[tm][0]);
            accd[tm][1] = wmma_bf16(a, bd1, accd[tm][1]);
        }
    }

    const int mo = (lane >> 4) << 3, no = lane & 15;
#pragma unroll
    for (int tm = 0; tm < 4; ++tm)
#pragma unroll
        for (int tn = 0; tn < 2; ++tn) {
            const size_t col = (size_t)(N0 + wn + 16 * tn + no);
#pragma unroll
            for (int e = 0; e < 8; ++e) {
                size_t tok = (bh << 7) + wm + 16 * tm + mo + e;
                float sq = (float)sqb[(tok << 9) + col];
                float y  = sq * accn[tm][tn][e] / accd[tm][tn][e];
                yb[(tok << 9) + col] = (bf16)y;
            }
        }
}

// ---------------------------------------------------------------------------
// Kernel 5: out = y @ Wo + bo  (f32 output)
// ---------------------------------------------------------------------------
__global__ __launch_bounds__(256) void out_kernel(
        const bf16* __restrict__ yb, const bf16* __restrict__ WoT,
        const float* __restrict__ bo, float* __restrict__ out) {
    extern __shared__ char smem[];
    bf16* At  = (bf16*)smem;
    bf16* Bt0 = At + 128 * LDA_Q;
    bf16* Bt1 = Bt0 + 128 * LDB_Q;

    const int tid = threadIdx.x, lane = tid & 31, wave = tid >> 5;
    const int wm = (wave >> 2) * 64, wn = (wave & 3) * 32;
    const int N0 = blockIdx.x * 128;
    const size_t M0 = (size_t)blockIdx.y * 128;

    for (int i = tid; i < 8192; i += 256) {
        int r = i >> 6, c = i & 63;
        async_ld_b128(lds_off_u32(At + r * LDA_Q + c * 8),
                      yb + ((M0 + r) << 9) + c * 8);
    }

    v8f acc[4][2];
#pragma unroll
    for (int tm = 0; tm < 4; ++tm) { acc[tm][0] = vzero8(); acc[tm][1] = vzero8(); }
    gemm_pass512(At, Bt0, Bt1, WoT, N0, lane, wave, wm, wn, acc);

    const int mo = (lane >> 4) << 3, no = lane & 15;
#pragma unroll
    for (int tm = 0; tm < 4; ++tm)
#pragma unroll
        for (int tn = 0; tn < 2; ++tn) {
            const size_t col = (size_t)(N0 + wn + 16 * tn + no);
            const float bias = bo[col];
#pragma unroll
            for (int e = 0; e < 8; ++e) {
                size_t row = M0 + wm + 16 * tm + mo + e;
                out[(row << 9) + col] = acc[tm][tn][e] + bias;
            }
        }
}

// ---------------------------------------------------------------------------
extern "C" void kernel_launch(void* const* d_in, const int* in_sizes, int n_in,
                              void* d_out, int out_size, void* d_ws, size_t ws_size,
                              hipStream_t stream) {
    const float* x    = (const float*)d_in[0];
    const float* Wq   = (const float*)d_in[1];
    const float* bq   = (const float*)d_in[2];
    const float* Wk   = (const float*)d_in[3];
    const float* bk   = (const float*)d_in[4];
    const float* Wv   = (const float*)d_in[5];
    const float* bv   = (const float*)d_in[6];
    const float* wtab = (const float*)d_in[7];
    const float* Wo   = (const float*)d_in[8];
    const float* bo   = (const float*)d_in[9];
    float* out = (float*)d_out;

    char* ws = (char*)d_ws;
    const size_t MC = (size_t)MTOT * C_;           // 67,108,864 elems
    size_t o = 0;
    bf16* xb   = (bf16*)(ws + o); o += MC * 2;     // reused as yb after qkv
    bf16* sqb  = (bf16*)(ws + o); o += MC * 2;
    bf16* ekb  = (bf16*)(ws + o); o += MC * 2;
    bf16* ekvb = (bf16*)(ws + o); o += MC * 2;
    bf16* WT   = (bf16*)(ws + o); o += (size_t)4 * 512 * 512 * 2;
    bf16* ewb  = (bf16*)(ws + o); o += 128 * 128 * 2;
    bf16* yb   = xb;   // x dead after qkv_kernel; alias saves 134 MB

    cvt_x_kernel<<<8192, 256, 0, stream>>>(x, xb, (int)(MC / 4));
    cvt_w_kernel<<<2080, 256, 0, stream>>>(Wq, Wk, Wv, Wo, wtab, WT, ewb);

    const size_t smem_gemm =
        (size_t)(128 * LDA_Q + 2 * 128 * LDB_Q) * 2;   // 169,984 B
    qkv_kernel<<<dim3(4, 1024), 256, smem_gemm, stream>>>(
        xb, WT + 1 * 262144, WT + 2 * 262144, WT + 0 * 262144,
        bk, bv, bq, ekb, ekvb, sqb);

    const size_t smem_aft = (size_t)3 * 128 * LDA_A * 2;   // 104,448 B
    aft_kernel<<<dim3(4, 1024), 256, smem_aft, stream>>>(ewb, ekb, ekvb, sqb, yb);

    out_kernel<<<dim3(4, 1024), 256, smem_gemm, stream>>>(
        yb, WT + 3 * 262144, bo, out);
}
